// ChampferLoss_18992345383226
// MI455X (gfx1250) — matrix operand order repaired and checked by reading, not compiled, not run on hardware
//
#include <hip/hip_runtime.h>

// ---- CDNA5 WMMA types ----
typedef __attribute__((ext_vector_type(16))) __bf16 v16bf;
typedef __attribute__((ext_vector_type(8)))  __bf16 v8bf;
typedef __attribute__((ext_vector_type(8)))  float  v8f;

#define WMMA_BF16(A, B, C) \
  __builtin_amdgcn_wmma_f32_16x16x32_bf16(false, (A), false, (B), (short)0, (C), false, false)

constexpr int   kB   = 32;     // batches
constexpr int   kN   = 2048;   // points per side (N == M)
constexpr int   kD   = 128;    // feature dim
constexpr int   BLK  = 128;    // row-block / col-step size
constexpr int   LDT  = 136;    // padded LDS stride in bf16 elems (272B = 68 dwords -> 4-bank stagger)
constexpr float PADV = 1000000.0f;
constexpr float PAD2 = PADV * PADV;   // masked squared distance

union FragU { v16bf v; v8bf h[2]; };

// Stage a contiguous 128x128 fp32 tile -> bf16 hi/lo in LDS.
// Row norms: each wave covers exactly one full row per iteration (r = wave + 8*it),
// so the norm is a full-wave butterfly sum -> lane 0 stores. No LDS atomics.
__device__ __forceinline__ void load_tile_bf16hl(const float* __restrict__ src,
                                                 __bf16* hi, __bf16* lo,
                                                 float* sq, int t)
{
  const int     lane = t & 31;
  const float4* p    = reinterpret_cast<const float4*>(src);
#pragma unroll
  for (int it = 0; it < (BLK * kD / 4) / 256; ++it) {   // 16 iters
    int    idx4 = t + it * 256;
    float4 v    = p[idx4];
    int    r    = idx4 >> 5;            // kD/4 = 32 float4 per row; r == (t>>5) + 8*it
    int    c    = (idx4 & 31) << 2;
    __bf16 hx = (__bf16)v.x, hy = (__bf16)v.y, hz = (__bf16)v.z, hw = (__bf16)v.w;
    __bf16* hp = hi + r * LDT + c;
    hp[0] = hx; hp[1] = hy; hp[2] = hz; hp[3] = hw;
    __bf16* lp = lo + r * LDT + c;
    lp[0] = (__bf16)(v.x - (float)hx);
    lp[1] = (__bf16)(v.y - (float)hy);
    lp[2] = (__bf16)(v.z - (float)hz);
    lp[3] = (__bf16)(v.w - (float)hw);
    float s = v.x * v.x + v.y * v.y + v.z * v.z + v.w * v.w;
    s += __shfl_xor(s, 1, 32);
    s += __shfl_xor(s, 2, 32);
    s += __shfl_xor(s, 4, 32);
    s += __shfl_xor(s, 8, 32);
    s += __shfl_xor(s, 16, 32);
    if (lane == 0) sq[r] = s;
  }
}

// One direction of the Chamfer loss:
//   for each row r in `rows`: min over cols of masked squared distance, * rowW,
//   summed over the workgroup's 128 rows -> partial[batch*16 + rowblock].
__global__ __launch_bounds__(256)
void chamfer_half(const float* __restrict__ rows, const float* __restrict__ rowW,
                  const float* __restrict__ cols, const float* __restrict__ colW,
                  float* __restrict__ partial)
{
  __shared__ __align__(16) __bf16 sAhi[BLK * LDT];
  __shared__ __align__(16) __bf16 sAlo[BLK * LDT];
  __shared__ __align__(16) __bf16 sBhi[BLK * LDT];
  __shared__ __align__(16) __bf16 sBlo[BLK * LDT];
  __shared__ float sSqA[BLK], sSqB[BLK], sWA[BLK], sWB[BLK];
  __shared__ float sWavePart[8];

  const int t     = threadIdx.x;
  const int lane  = t & 31;          // wave32
  const int wave  = t >> 5;          // 8 waves
  const int half  = lane >> 4;
  const int l15   = lane & 15;
  const int rblk  = blockIdx.x;      // 0..15
  const int batch = blockIdx.y;      // 0..31
  const int rbase = rblk * BLK;

  // ---- stage A (this block's 128 rows), row norms, row weights ----
  if (t < BLK) sWA[t] = rowW[batch * kN + rbase + t];
  load_tile_bf16hl(rows + (size_t)(batch * kN + rbase) * kD, sAhi, sAlo, sSqA, t);
  __syncthreads();

  const int wrow = wave * 16;        // this wave's 16-row strip
  float sqa_r[8];
#pragma unroll
  for (int r = 0; r < 8; ++r) sqa_r[r] = sSqA[wrow + r + 8 * half];

  float rmin[8];
#pragma unroll
  for (int r = 0; r < 8; ++r) rmin[r] = PAD2;

  // ---- sweep all columns in 128-wide steps ----
  for (int cs = 0; cs < kN / BLK; ++cs) {
    __syncthreads();   // previous step's epilogue done before B overwrite
    if (t < BLK) sWB[t] = colW[batch * kN + cs * BLK + t];
    load_tile_bf16hl(cols + (size_t)(batch * kN + cs * BLK) * kD, sBhi, sBlo, sSqB, t);

    // prefetch next tile into WGP$/L2 while we compute this one
    if (cs + 1 < kN / BLK) {
      const char* nxt = (const char*)(cols + (size_t)(batch * kN + (cs + 1) * BLK) * kD);
#pragma unroll
      for (int it = 0; it < (BLK * kD / 4) / 256; ++it)
        __builtin_prefetch(nxt + (size_t)(t + it * 256) * 16, 0, 3);
    }
    __syncthreads();

    v8f acc[8];
#pragma unroll
    for (int ct = 0; ct < 8; ++ct)
#pragma unroll
      for (int j = 0; j < 8; ++j) acc[ct][j] = 0.0f;

#pragma unroll
    for (int kc = 0; kc < kD / 32; ++kc) {
      const int k0 = kc * 32;
      // A fragment (16x32 bf16, contiguous-K layout k = 16*half + s), hi and lo
      FragU     ah, al;
      const int aoff = (wrow + l15) * LDT + k0 + 16 * half;
      ah.h[0] = *(const v8bf*)&sAhi[aoff];
      ah.h[1] = *(const v8bf*)&sAhi[aoff + 8];
      al.h[0] = *(const v8bf*)&sAlo[aoff];
      al.h[1] = *(const v8bf*)&sAlo[aoff + 8];
      // process col-tiles in pairs: two independent accumulator chains interleaved
#pragma unroll
      for (int ctp = 0; ctp < 4; ++ctp) {
        const int ct0 = ctp * 2, ct1 = ct0 + 1;
        const int b0  = (ct0 * 16 + l15) * LDT + k0 + 16 * half;
        const int b1  = (ct1 * 16 + l15) * LDT + k0 + 16 * half;
        FragU     bh0, bl0, bh1, bl1;
        bh0.h[0] = *(const v8bf*)&sBhi[b0];
        bh0.h[1] = *(const v8bf*)&sBhi[b0 + 8];
        bl0.h[0] = *(const v8bf*)&sBlo[b0];
        bl0.h[1] = *(const v8bf*)&sBlo[b0 + 8];
        bh1.h[0] = *(const v8bf*)&sBhi[b1];
        bh1.h[1] = *(const v8bf*)&sBhi[b1 + 8];
        bl1.h[0] = *(const v8bf*)&sBlo[b1];
        bl1.h[1] = *(const v8bf*)&sBlo[b1 + 8];
        // split-bf16 fp32 recovery: hi*hi + hi*lo + lo*hi (interleaved across ct0/ct1)
        acc[ct0] = WMMA_BF16(ah.v, bh0.v, acc[ct0]);
        acc[ct1] = WMMA_BF16(ah.v, bh1.v, acc[ct1]);
        acc[ct0] = WMMA_BF16(ah.v, bl0.v, acc[ct0]);
        acc[ct1] = WMMA_BF16(ah.v, bl1.v, acc[ct1]);
        acc[ct0] = WMMA_BF16(al.v, bh0.v, acc[ct0]);
        acc[ct1] = WMMA_BF16(al.v, bh1.v, acc[ct1]);
      }
    }

    // fold tile into running per-row min of squared distances
#pragma unroll
    for (int ct = 0; ct < 8; ++ct) {
      float wb  = sWB[ct * 16 + l15];
      float sqb = sSqB[ct * 16 + l15];
      bool  bv  = wb > 0.f;
#pragma unroll
      for (int r = 0; r < 8; ++r) {
        float sq = sqa_r[r] + sqb - 2.0f * acc[ct][r];
        sq       = fmaxf(sq, 1e-12f);
        rmin[r]  = fminf(rmin[r], bv ? sq : PAD2);
      }
    }
  }

  // ---- reduce: cross-lane min within 16-lane halves, weight, sum ----
  float sum = 0.f;
#pragma unroll
  for (int r = 0; r < 8; ++r) {
    float m = rmin[r];
    m = fminf(m, __shfl_xor(m, 1, 32));
    m = fminf(m, __shfl_xor(m, 2, 32));
    m = fminf(m, __shfl_xor(m, 4, 32));
    m = fminf(m, __shfl_xor(m, 8, 32));
    float w = sWA[wrow + r + 8 * half];
    if (!(w > 0.f)) m = PAD2;          // fully-masked row: min dist == PAD
    sum += m * w;
  }
  sum += __shfl_xor(sum, 16, 32);      // combine the two halves
  if (lane == 0) sWavePart[wave] = sum;
  __syncthreads();
  if (t == 0) {
    float s = 0.f;
#pragma unroll
    for (int w = 0; w < 8; ++w) s += sWavePart[w];
    partial[batch * (kN / BLK) + rblk] = s;   // fixed slot: deterministic
  }
}

__global__ void chamfer_reduce(const float* __restrict__ partial, float* __restrict__ out)
{
  int b = threadIdx.x;
  if (b < kB) {
    float s = 0.f;
    for (int i = 0; i < kN / BLK; ++i) s += partial[b * (kN / BLK) + i];                       // pass 0
    for (int i = 0; i < kN / BLK; ++i) s += partial[kB * (kN / BLK) + b * (kN / BLK) + i];     // pass 1
    out[b] = s;
  }
}

extern "C" void kernel_launch(void* const* d_in, const int* in_sizes, int n_in,
                              void* d_out, int out_size, void* d_ws, size_t ws_size,
                              hipStream_t stream)
{
  (void)in_sizes; (void)n_in; (void)out_size; (void)ws_size;
  const float* oW      = (const float*)d_in[0];   // [B,N]
  const float* outputs = (const float*)d_in[1];   // [B,N,D]
  const float* tW      = (const float*)d_in[2];   // [B,M]
  const float* targets = (const float*)d_in[3];   // [B,M,D]
  float*       partial = (float*)d_ws;            // 2 * 32 * 16 floats = 4 KB
  float*       out     = (float*)d_out;           // [B]

  dim3 grid(kN / BLK, kB), block(256);
  // pass 0: rows = outputs  -> sum_n min_m
  chamfer_half<<<grid, block, 0, stream>>>(outputs, oW, targets, tW, partial);
  // pass 1: rows = targets  -> sum_m min_n
  chamfer_half<<<grid, block, 0, stream>>>(targets, tW, outputs, oW, partial + kB * (kN / BLK));
  chamfer_reduce<<<1, kB, 0, stream>>>(partial, out);
}
